// BCDReverseTransform_38929583571136
// MI455X (gfx1250) — compile-verified, matrix operand order checked by reading, not run on hardware
//
#include <hip/hip_runtime.h>

// BCD reverse transform: out[i] = sum_j ((sign(x[i,j])+1)/2) * 2^j
// N = 4M rows, D = 16. Pure streaming problem (256 MB in / 16 MB out,
// ~11.7 us floor at 23.3 TB/s). Dot products done on the WMMA path:
// one v_wmma_f32_16x16x32_f16 per 32 rows per wave.

typedef __attribute__((ext_vector_type(16))) _Float16 v16h;
typedef __attribute__((ext_vector_type(8)))  float    v8f;
typedef __attribute__((ext_vector_type(4)))  float    v4f;

__device__ __forceinline__ _Float16 bit_of(float x) {
    // (sign(x)+1)*0.5 : x>0 -> 1, x==0 (incl -0) -> 0.5, x<0 -> 0
    float t = (x > 0.0f) ? 1.0f : ((x == 0.0f) ? 0.5f : 0.0f);
    return (_Float16)t;
}

__global__ __launch_bounds__(256) void bcd_wmma_kernel(const float* __restrict__ x,
                                                       float* __restrict__ out,
                                                       int nChunks /* = N/32 */) {
    const int lane       = threadIdx.x & 31;
    const int waveInBlk  = threadIdx.x >> 5;
    const int wavesPerBlk = blockDim.x >> 5;

    const int m     = lane & 15;   // row-in-group / output column id
    const int khalf = lane >> 4;   // 0: K 0-7 & 16-23, 1: K 8-15 & 24-31

    // ---- Build B (32x16 f16) once per wave ----------------------------------
    // B[k,0] = 2^k        for k in [0,16)
    // B[k,1] = 2^(k-16)   for k in [16,32)
    // all other entries 0.
    // 16-bit B layout: lane (0-15) = column n=lane, elems 0..7 -> K=khalf*8+e,
    // elems 8..15 -> K=16+khalf*8+(e-8); lane+16 same column, upper K halves.
    v16h B;
#pragma unroll
    for (int e = 0; e < 16; ++e) {
        const int K = (e < 8) ? (khalf * 8 + e) : (16 + khalf * 8 + (e - 8));
        float w = 0.0f;
        if (K < 16) { if (m == 0) w = (float)(1u << K); }
        else        { if (m == 1) w = (float)(1u << (K - 16)); }
        B[e] = (_Float16)w;
    }

    int chunk = blockIdx.x * wavesPerBlk + waveInBlk;
    const int stride = gridDim.x * wavesPerBlk;

    for (; chunk < nChunks; chunk += stride) {   // wave-uniform: EXEC all ones at WMMA
        const long long r0 = (long long)chunk * 32;

        // lane m   : floats 0-7  of rows r0+m and r0+16+m
        // lane m+16: floats 8-15 of rows r0+m and r0+16+m
        const float* pa = x + (r0 + m)      * 16 + khalf * 8;
        const float* pb = x + (r0 + 16 + m) * 16 + khalf * 8;

        const v4f a0 = __builtin_nontemporal_load((const v4f*)pa);
        const v4f a1 = __builtin_nontemporal_load((const v4f*)(pa + 4));
        const v4f a2 = __builtin_nontemporal_load((const v4f*)pb);
        const v4f a3 = __builtin_nontemporal_load((const v4f*)(pb + 4));

        // A (16x32 f16): elems 0-7 = bits of row r0+m slice, 8-15 = row r0+16+m slice
        v16h A;
#pragma unroll
        for (int e = 0; e < 4; ++e) {
            A[e]      = bit_of(a0[e]);
            A[e + 4]  = bit_of(a1[e]);
            A[e + 8]  = bit_of(a2[e]);
            A[e + 12] = bit_of(a3[e]);
        }

        v8f c = {0.f, 0.f, 0.f, 0.f, 0.f, 0.f, 0.f, 0.f};
        // D = A * B + 0 : D[:,0] = out rows r0..r0+15, D[:,1] = rows r0+16..r0+31
        c = __builtin_amdgcn_wmma_f32_16x16x32_f16(
                /*neg_a=*/false, A, /*neg_b=*/false, B,
                /*c_mod=*/(short)0, c, /*reuse_a=*/false, /*reuse_b=*/false);

        // C/D f32 layout: lane n (0-15) col n rows 0-7 in c[0..7];
        // lane n+16 col n rows 8-15. Columns 0,1 hold results:
        //   lane 0 -> out[r0+0..7],  lane 16 -> out[r0+8..15]
        //   lane 1 -> out[r0+16..23], lane 17 -> out[r0+24..31]
        if (m < 2) {
            float* po = out + r0 + (long long)m * 16 + khalf * 8;
            const v4f lo = {c[0], c[1], c[2], c[3]};
            const v4f hi = {c[4], c[5], c[6], c[7]};
            __builtin_nontemporal_store(lo, (v4f*)po);
            __builtin_nontemporal_store(hi, (v4f*)(po + 4));
        }
    }
}

// Scalar tail for N % 32 rows (not used for N=4M, kept for generality).
__global__ void bcd_tail_kernel(const float* __restrict__ x,
                                float* __restrict__ out,
                                long long startRow, long long N) {
    long long r = startRow + (long long)blockIdx.x * blockDim.x + threadIdx.x;
    if (r >= N) return;
    const float* p = x + r * 16;
    float acc = 0.0f, w = 1.0f;
#pragma unroll
    for (int j = 0; j < 16; ++j) {
        float v = p[j];
        float b = (v > 0.0f) ? 1.0f : ((v == 0.0f) ? 0.5f : 0.0f);
        acc += b * w;
        w *= 2.0f;
    }
    out[r] = acc;
}

extern "C" void kernel_launch(void* const* d_in, const int* in_sizes, int n_in,
                              void* d_out, int out_size, void* d_ws, size_t ws_size,
                              hipStream_t stream) {
    const float* x   = (const float*)d_in[0];
    float*       out = (float*)d_out;

    const long long total = (long long)in_sizes[0]; // N * 16
    const long long N     = total / 16;
    const long long nChunks = N / 32;               // 32 rows per wave-chunk

    if (nChunks > 0) {
        const int wavesPerBlk   = 8;   // 256 threads
        const int chunksPerWave = 4;   // amortize B setup, pipeline loads
        long long blocks = (nChunks + (long long)wavesPerBlk * chunksPerWave - 1) /
                           ((long long)wavesPerBlk * chunksPerWave);
        if (blocks < 1) blocks = 1;
        bcd_wmma_kernel<<<(int)blocks, 256, 0, stream>>>(x, out, (int)nChunks);
    }

    const long long tail = N - nChunks * 32;
    if (tail > 0) {
        bcd_tail_kernel<<<(int)((tail + 255) / 256), 256, 0, stream>>>(
            x, out, nChunks * 32, N);
    }
}
